// GTEProgramClassification_27986006900815
// MI455X (gfx1250) — matrix operand order repaired and checked by reading, not compiled, not run on hardware
//
#include <hip/hip_runtime.h>

typedef unsigned short u16;
typedef __bf16  v4bf  __attribute__((ext_vector_type(4)));
typedef __bf16  v8bf  __attribute__((ext_vector_type(8)));
typedef __bf16  v16bf __attribute__((ext_vector_type(16)));
typedef float   v8f   __attribute__((ext_vector_type(8)));

#define NDSTn 8192
#define Dn    512
#define Sn    8
#define Hn    8
#define ROWS  (Sn * NDSTn)   /* 65536 */

__device__ __forceinline__ u16 f2b(float f) {
    return __builtin_bit_cast(u16, (__bf16)f);   // native v_cvt bf16
}
__device__ __forceinline__ float b2f(unsigned int h) {
    return __uint_as_float(h << 16);             // single v_lshlrev
}

// ---------------------------------------------------------------------------
// f32 -> bf16 conversion (weights)
// ---------------------------------------------------------------------------
__global__ __launch_bounds__(256) void f2bf_kernel(const float* __restrict__ src,
                                                   u16* __restrict__ dst, int n) {
    int i = blockIdx.x * 256 + threadIdx.x;
    if (i < n) dst[i] = f2b(src[i]);
}

// ---------------------------------------------------------------------------
// Gather: x[s*N+n][d] = emb[token_ids[edge_src[n][s]]][d]   (s in 0..7 of 9)
// ---------------------------------------------------------------------------
__global__ __launch_bounds__(256) void gather_kernel(const int* __restrict__ token_ids,
                                                     const int* __restrict__ edge_src,
                                                     const float* __restrict__ emb,
                                                     float* __restrict__ xf) {
    const int row = blockIdx.x;            // row = s*NDST + n
    const int s = row >> 13;
    const int n = row & (NDSTn - 1);
    const int src = edge_src[n * 9 + s];
    const int tok = token_ids[src];
    const int tid = threadIdx.x;
    const size_t db = (size_t)row * Dn;
    const size_t sb = (size_t)tok * Dn;
    xf[db + tid]       = emb[sb + tid];
    xf[db + tid + 256] = emb[sb + tid + 256];
}

// ---------------------------------------------------------------------------
// Generic tiled bf16 WMMA GEMM: Out[M,N] = A[M,K] * W[N,K]^T + bias
//   block tile 128x128, BK=32, 8 waves, each wave -> 32x64 (2x4 WMMA frags)
//   register-staged double buffer: tile k+1 global loads overlap tile k math
// ---------------------------------------------------------------------------
#define BM 128
#define BN 128
#define BK 32
#define LSTR 40   // LDS row stride in bf16 halves (padded)

template <bool A_F32, bool OUT_BF16, bool RELU>
__global__ __launch_bounds__(256) void wmma_gemm_kernel(
        const void* __restrict__ Aptr, const u16* __restrict__ Bw,
        const float* __restrict__ bias, void* __restrict__ Out,
        int K, int Nbound, int ldo) {
    __shared__ __align__(16) u16 lsA[BM * LSTR];
    __shared__ __align__(16) u16 lsB[BN * LSTR];

    const int tid  = threadIdx.x;
    const int lane = tid & 31;
    const int wid  = tid >> 5;
    const int wm   = (wid & 3) * 32;      // wave row offset in tile
    const int wn   = (wid >> 2) * 64;     // wave col offset in tile
    const int fr   = lane & 15;
    const int kh   = lane >> 4;
    const int row0 = blockIdx.y * BM;
    const int col0 = blockIdx.x * BN;

    // ---- per-thread staging pointers (bumped by BK each tile) ----
    // A (f32): 8 threads cover 32 cols (4 each), 32 rows/pass, 4 passes
    // A (bf16): 4 threads cover 32 cols (8 each), 64 rows/pass, 2 passes
    const int ac4 = (tid & 7) * 4;
    const int arb = tid >> 3;
    const int ac8 = (tid & 3) * 8;
    const int arb2 = tid >> 2;
    const float* aP[4];
    const u16*   aPB[2];
    if constexpr (A_F32) {
#pragma unroll
        for (int p = 0; p < 4; ++p)
            aP[p] = (const float*)Aptr + (size_t)(row0 + arb + p * 32) * K + ac4;
    } else {
#pragma unroll
        for (int p = 0; p < 2; ++p)
            aPB[p] = (const u16*)Aptr + (size_t)(row0 + arb2 + p * 64) * K + ac8;
    }
    // B: 4 threads cover 32 cols (8 each), 64 n-rows/pass, 2 passes
    const u16* bP[2];
    bool bOk[2];
#pragma unroll
    for (int p = 0; p < 2; ++p) {
        const int ng = col0 + arb2 + p * 64;
        bOk[p] = ng < Nbound;
        bP[p] = Bw + (size_t)ng * K + ac8;
    }

    v8f acc[2][4];
#pragma unroll
    for (int i = 0; i < 2; ++i)
#pragma unroll
        for (int j = 0; j < 4; ++j)
#pragma unroll
            for (int v = 0; v < 8; ++v) acc[i][j][v] = 0.0f;

    // ---- staged registers ----
    float4 ra[4];
    uint4  rab[2];
    uint4  rb[2];

    auto loadTiles = [&]() {
        if constexpr (A_F32) {
#pragma unroll
            for (int p = 0; p < 4; ++p) { ra[p] = *(const float4*)aP[p]; aP[p] += BK; }
        } else {
#pragma unroll
            for (int p = 0; p < 2; ++p) { rab[p] = *(const uint4*)aPB[p]; aPB[p] += BK; }
        }
#pragma unroll
        for (int p = 0; p < 2; ++p) {
            uint4 v; v.x = 0; v.y = 0; v.z = 0; v.w = 0;
            if (bOk[p]) v = *(const uint4*)bP[p];
            bP[p] += BK;
            rb[p] = v;
        }
    };

    loadTiles();   // tile 0

    for (int kt = 0; kt < K; kt += BK) {
        // ---- store staged tile to LDS (converting A to bf16 if needed) ----
        if constexpr (A_F32) {
#pragma unroll
            for (int p = 0; p < 4; ++p) {
                v4bf c;
                c[0] = (__bf16)ra[p].x; c[1] = (__bf16)ra[p].y;
                c[2] = (__bf16)ra[p].z; c[3] = (__bf16)ra[p].w;
                *(v4bf*)&lsA[(arb + p * 32) * LSTR + ac4] = c;
            }
        } else {
#pragma unroll
            for (int p = 0; p < 2; ++p)
                *(uint4*)&lsA[(arb2 + p * 64) * LSTR + ac8] = rab[p];
        }
#pragma unroll
        for (int p = 0; p < 2; ++p)
            *(uint4*)&lsB[(arb2 + p * 64) * LSTR + ac8] = rb[p];
        __syncthreads();

        // ---- issue next tile's global loads (latency hidden under math) ----
        if (kt + BK < K) {
            loadTiles();
            // prefetch one more tile ahead into L2/L0
            if (kt + 2 * BK < K) {
                if constexpr (A_F32) __builtin_prefetch(aP[0], 0, 0);
                else                 __builtin_prefetch(aPB[0], 0, 0);
                if (bOk[0]) __builtin_prefetch(bP[0], 0, 0);
            }
        }

        // ---- fragments from LDS ----
        v16bf af[2], bfr[4];
#pragma unroll
        for (int i = 0; i < 2; ++i) {
            const int r = wm + 16 * i + fr;
            v8bf lo = *(const v8bf*)&lsA[r * LSTR + 8 * kh];
            v8bf hi = *(const v8bf*)&lsA[r * LSTR + 16 + 8 * kh];
            af[i] = __builtin_shufflevector(lo, hi, 0, 1, 2, 3, 4, 5, 6, 7,
                                            8, 9, 10, 11, 12, 13, 14, 15);
        }
#pragma unroll
        for (int j = 0; j < 4; ++j) {
            const int nn = wn + 16 * j + fr;
            v8bf lo = *(const v8bf*)&lsB[nn * LSTR + 8 * kh];
            v8bf hi = *(const v8bf*)&lsB[nn * LSTR + 16 + 8 * kh];
            bfr[j] = __builtin_shufflevector(lo, hi, 0, 1, 2, 3, 4, 5, 6, 7,
                                             8, 9, 10, 11, 12, 13, 14, 15);
        }
        // ---- 8 WMMAs ----
#pragma unroll
        for (int i = 0; i < 2; ++i)
#pragma unroll
            for (int j = 0; j < 4; ++j)
                acc[i][j] = __builtin_amdgcn_wmma_f32_16x16x32_bf16(
                    false, af[i], false, bfr[j], (short)0, acc[i][j], false, false);
        __syncthreads();
    }

    // ---- epilogue: bias (+relu), store f32 or bf16 ----
#pragma unroll
    for (int j = 0; j < 4; ++j) {
        const int col = col0 + wn + 16 * j + fr;
        if (col < Nbound) {
            const float bv = bias[col];
#pragma unroll
            for (int i = 0; i < 2; ++i) {
#pragma unroll
                for (int v = 0; v < 8; ++v) {
                    const int rowg = row0 + wm + 16 * i + 8 * kh + v;
                    float val = acc[i][j][v] + bv;
                    if (RELU) val = fmaxf(val, 0.0f);
                    if (OUT_BF16)
                        ((u16*)Out)[(size_t)rowg * ldo + col] = f2b(val);
                    else
                        ((float*)Out)[(size_t)rowg * ldo + col] = val;
                }
            }
        }
    }
}

// ---------------------------------------------------------------------------
// Attention: per (n,h,i): scores over 8 keys, softmax, weighted sum of v.
// qkv: bf16 [S*N][1536] (q|k|v), out: bf16 [S*N][512]
// ---------------------------------------------------------------------------
__global__ __launch_bounds__(256) void attn_kernel(const u16* __restrict__ qkv,
                                                   u16* __restrict__ aout) {
    const int gid = blockIdx.x * 256 + threadIdx.x;   // NDST*H*S = 524288
    const int i = gid & 7;
    const int h = (gid >> 3) & 7;
    const int n = gid >> 6;
    const unsigned int hoff = h * 64;

    const unsigned int* qp =
        (const unsigned int*)(qkv + ((size_t)i * NDSTn + n) * 1536 + hoff);
    float q[64];
#pragma unroll
    for (int c = 0; c < 32; ++c) {
        const unsigned int u = qp[c];
        q[2 * c]     = b2f(u & 0xffffu);
        q[2 * c + 1] = b2f(u >> 16);
    }
    float sc[8];
    float mx = -3.0e38f;
#pragma unroll
    for (int j = 0; j < 8; ++j) {
        const unsigned int* kp =
            (const unsigned int*)(qkv + ((size_t)j * NDSTn + n) * 1536 + 512 + hoff);
        float d = 0.0f;
#pragma unroll
        for (int c = 0; c < 32; ++c) {
            const unsigned int u = kp[c];
            d += q[2 * c] * b2f(u & 0xffffu) + q[2 * c + 1] * b2f(u >> 16);
        }
        sc[j] = d * 0.125f;            // 1/sqrt(64)
        mx = fmaxf(mx, sc[j]);
    }
    float se = 0.0f;
#pragma unroll
    for (int j = 0; j < 8; ++j) { sc[j] = __expf(sc[j] - mx); se += sc[j]; }
    const float inv = 1.0f / se;
    float acc[64];
#pragma unroll
    for (int c = 0; c < 64; ++c) acc[c] = 0.0f;
#pragma unroll
    for (int j = 0; j < 8; ++j) {
        const float w = sc[j] * inv;
        const unsigned int* vp =
            (const unsigned int*)(qkv + ((size_t)j * NDSTn + n) * 1536 + 1024 + hoff);
#pragma unroll
        for (int c = 0; c < 32; ++c) {
            const unsigned int u = vp[c];
            acc[2 * c]     += w * b2f(u & 0xffffu);
            acc[2 * c + 1] += w * b2f(u >> 16);
        }
    }
    u16* op = aout + ((size_t)i * NDSTn + n) * Dn + hoff;
#pragma unroll
    for (int c = 0; c < 64; ++c) op[c] = f2b(acc[c]);
}

// ---------------------------------------------------------------------------
// Residual add + LayerNorm (in place on xf).  One block per row of 512.
// ---------------------------------------------------------------------------
__global__ __launch_bounds__(256) void add_ln_kernel(float* __restrict__ xf,
                                                     const float* __restrict__ g,
                                                     const float* __restrict__ gamma,
                                                     const float* __restrict__ beta) {
    __shared__ float rs[256], rq[256];
    const int row = blockIdx.x;
    const int tid = threadIdx.x;
    const size_t base = (size_t)row * Dn;
    const float y0 = xf[base + tid] + g[base + tid];
    const float y1 = xf[base + tid + 256] + g[base + tid + 256];
    rs[tid] = y0 + y1;
    rq[tid] = y0 * y0 + y1 * y1;
    __syncthreads();
    for (int st = 128; st > 0; st >>= 1) {
        if (tid < st) { rs[tid] += rs[tid + st]; rq[tid] += rq[tid + st]; }
        __syncthreads();
    }
    const float mean = rs[0] * (1.0f / 512.0f);
    const float var  = rq[0] * (1.0f / 512.0f) - mean * mean;
    const float rstd = rsqrtf(var + 1e-5f);
    xf[base + tid]       = (y0 - mean) * rstd * gamma[tid] + beta[tid];
    xf[base + tid + 256] = (y1 - mean) * rstd * gamma[tid + 256] + beta[tid + 256];
}

// ---------------------------------------------------------------------------
// Max-pool over S=8, output bf16 [NDST][D]
// ---------------------------------------------------------------------------
__global__ __launch_bounds__(256) void maxpool_kernel(const float* __restrict__ xf,
                                                      u16* __restrict__ rst) {
    const int idx = blockIdx.x * 256 + threadIdx.x;   // n*512 + d
    float m = xf[idx];
#pragma unroll
    for (int s = 1; s < 8; ++s)
        m = fmaxf(m, xf[(size_t)s * (NDSTn * Dn) + idx]);
    rst[idx] = f2b(m);
}

// ---------------------------------------------------------------------------
// Launch
// ---------------------------------------------------------------------------
extern "C" void kernel_launch(void* const* d_in, const int* in_sizes, int n_in,
                              void* d_out, int out_size, void* d_ws, size_t ws_size,
                              hipStream_t stream) {
    (void)in_sizes; (void)n_in; (void)out_size; (void)ws_size;
    const int*   token_ids = (const int*)d_in[0];
    const int*   edge_src  = (const int*)d_in[1];
    const float* emb   = (const float*)d_in[2];
    const float* Wqkv  = (const float*)d_in[3];
    const float* bqkv  = (const float*)d_in[4];
    const float* Wo    = (const float*)d_in[5];
    const float* bo    = (const float*)d_in[6];
    const float* W1    = (const float*)d_in[7];
    const float* b1    = (const float*)d_in[8];
    const float* W2    = (const float*)d_in[9];
    const float* b2    = (const float*)d_in[10];
    const float* ln1g  = (const float*)d_in[11];
    const float* ln1b  = (const float*)d_in[12];
    const float* ln2g  = (const float*)d_in[13];
    const float* ln2b  = (const float*)d_in[14];
    const float* Wfc   = (const float*)d_in[15];
    const float* bfc   = (const float*)d_in[16];
    float* out = (float*)d_out;

    // workspace layout
    const size_t XF_B  = (size_t)ROWS * Dn * 4;      // 134,217,728  f32 residual stream
    const size_t QKV_B = (size_t)ROWS * 1536 * 2;    // 201,326,592  bf16 qkv / ffn hidden
    const size_t AB_B  = (size_t)ROWS * Dn * 2;      //  67,108,864  bf16 attn-out / pooled
    const size_t GO_B  = (size_t)ROWS * Dn * 4;      // 134,217,728  f32 pre-LN gemm out
    char* ws = (char*)d_ws;
    float* xf   = (float*)ws;
    u16*   qkvb = (u16*)(ws + XF_B);
    u16*   abuf = (u16*)(ws + XF_B + QKV_B);
    float* gout = (float*)(ws + XF_B + QKV_B + AB_B);
    u16*   wb   = (u16*)(ws + XF_B + QKV_B + AB_B + GO_B);
    u16* WqkvB = wb;                        // 2*1536*512 = 1,572,864
    u16* WoB   = WqkvB + 1572864;           // 2*512*512  =   524,288
    u16* W1B   = WoB   + 524288;            // 2*1024*512 = 1,048,576
    u16* W2B   = W1B   + 1048576;           // 2*512*1024 = 1,048,576
    u16* WfcB  = W2B   + 1048576;           // 104*512    =    53,248

    // weight conversion to bf16
    f2bf_kernel<<<(1572864 + 255) / 256, 256, 0, stream>>>(Wqkv, WqkvB, 1572864);
    f2bf_kernel<<<(524288  + 255) / 256, 256, 0, stream>>>(Wo,   WoB,   524288);
    f2bf_kernel<<<(1048576 + 255) / 256, 256, 0, stream>>>(W1,   W1B,   1048576);
    f2bf_kernel<<<(1048576 + 255) / 256, 256, 0, stream>>>(W2,   W2B,   1048576);
    f2bf_kernel<<<(53248   + 255) / 256, 256, 0, stream>>>(Wfc,  WfcB,  53248);

    // embedding + mailbox gather  -> xf [65536,512]
    gather_kernel<<<ROWS, 256, 0, stream>>>(token_ids, edge_src, emb, xf);

    for (int l = 0; l < 2; ++l) {
        // qkv = x @ Wqkv^T + bqkv -> bf16 [65536,1536]
        wmma_gemm_kernel<true, true, false><<<dim3(12, ROWS / 128), 256, 0, stream>>>(
            xf, WqkvB + (size_t)l * 1536 * 512, bqkv + l * 1536, qkvb, 512, 1536, 1536);
        // attention per (n,h) -> bf16 [65536,512]
        attn_kernel<<<(NDSTn * Hn * Sn) / 256, 256, 0, stream>>>(qkvb, abuf);
        // a @ Wo^T + bo -> f32 [65536,512]
        wmma_gemm_kernel<false, false, false><<<dim3(4, ROWS / 128), 256, 0, stream>>>(
            abuf, WoB + (size_t)l * 512 * 512, bo + l * 512, gout, 512, 512, 512);
        // x = LN1(x + a)
        add_ln_kernel<<<ROWS, 256, 0, stream>>>(xf, gout, ln1g + l * 512, ln1b + l * 512);
        // h = relu(x @ W1^T + b1) -> bf16 [65536,1024] (reuse qkv buffer)
        wmma_gemm_kernel<true, true, true><<<dim3(8, ROWS / 128), 256, 0, stream>>>(
            xf, W1B + (size_t)l * 1024 * 512, b1 + l * 1024, qkvb, 512, 1024, 1024);
        // f = h @ W2^T + b2 -> f32 [65536,512]
        wmma_gemm_kernel<false, false, false><<<dim3(4, ROWS / 128), 256, 0, stream>>>(
            qkvb, W2B + (size_t)l * 512 * 1024, b2 + l * 512, gout, 1024, 512, 512);
        // x = LN2(x + f)
        add_ln_kernel<<<ROWS, 256, 0, stream>>>(xf, gout, ln2g + l * 512, ln2b + l * 512);
    }

    // max over S -> bf16 [8192,512] (reuse abuf)
    maxpool_kernel<<<(NDSTn * Dn) / 256, 256, 0, stream>>>(xf, abuf);
    // logits = rst @ Wfc^T + bfc -> f32 [8192,104]
    wmma_gemm_kernel<false, false, false><<<dim3(1, NDSTn / 128), 256, 0, stream>>>(
        abuf, WfcB, bfc, out, 512, 104, 104);
}
